// MganMixer_85899346061
// MI455X (gfx1250) — matrix operand order repaired and checked by reading, not compiled, not run on hardware
//
#include <hip/hip_runtime.h>
#include <hip/hip_bf16.h>

#define NAG   32
#define DOBS  64
#define HIDN  64
#define SDIMX 256
#define NHEAD 4
#define HYPX  64
#define NEGV  -9999999.0f
#define LDF   72   // row stride in halfs for [32][64] buffers (144B, 16B aligned)
#define LDT   40   // row stride in halfs for [32][32] buffers (80B, 16B aligned)
#define WELEM (NHEAD * HIDN * 128)   // elements in one weight tensor (32768)

typedef _Float16 v16h __attribute__((ext_vector_type(16)));
typedef _Float16 v8h  __attribute__((ext_vector_type(8)));
typedef float    v8f  __attribute__((ext_vector_type(8)));

#define WMMA(a, b, c) \
  __builtin_amdgcn_wmma_f32_16x16x32_f16(false, (a), false, (b), (short)0, (c), false, false)

// A fragment (16x32 f16): rows m0..m0+15, cols k0..k0+31 of row-major src.
// Lanes 0-15: M=lane, K = k0+[0..15]; lanes 16-31: same M, K = k0+[16..31].
__device__ __forceinline__ v16h frag_A(const _Float16* src, int ld, int m0, int k0, int lane) {
  int row = m0 + (lane & 15);
  int kb  = k0 + ((lane >> 4) << 4);
  const _Float16* p = src + row * ld + kb;
  v8h lo = *(const v8h*)(p);
  v8h hi = *(const v8h*)(p + 8);
  v16h r;
#pragma unroll
  for (int i = 0; i < 8; ++i) { r[i] = lo[i]; r[i + 8] = hi[i]; }
  return r;
}

// B fragment (32x16 f16) from an "n-major" source: src[n][k] contiguous in k.
__device__ __forceinline__ v16h frag_B_nm(const _Float16* src, int ld, int n0, int k0, int lane) {
  int row = n0 + (lane & 15);
  int kb  = k0 + ((lane >> 4) << 4);
  const _Float16* p = src + row * ld + kb;
  v8h lo = *(const v8h*)(p);
  v8h hi = *(const v8h*)(p + 8);
  v16h r;
#pragma unroll
  for (int i = 0; i < 8; ++i) { r[i] = lo[i]; r[i + 8] = hi[i]; }
  return r;
}

// B fragment (32x16 f16) from a k-major source: src[k][n] (strided per-lane gather).
__device__ __forceinline__ v16h frag_B_km(const _Float16* src, int ld, int n0, int k0, int lane) {
  int ncol = n0 + (lane & 15);
  int kb   = k0 + ((lane >> 4) << 4);
  v16h r;
#pragma unroll
  for (int i = 0; i < 16; ++i) r[i] = src[(kb + i) * ld + ncol];
  return r;
}

// B fragment of W^T from pre-converted f16 weights: B[k=c][n=e] = W16[e][c],
// W16 row-major [64][128] f16 in global (d_ws). 16 contiguous halfs per lane.
__device__ __forceinline__ v16h frag_B_wh(const _Float16* __restrict__ W, int n0, int k0,
                                          int lane) {
  int e  = n0 + (lane & 15);
  int kb = k0 + ((lane >> 4) << 4);
  const _Float16* p = W + e * 128 + kb;
  v8h lo = *(const v8h*)(p);
  v8h hi = *(const v8h*)(p + 8);
  v16h r;
#pragma unroll
  for (int i = 0; i < 8; ++i) { r[i] = lo[i]; r[i + 8] = hi[i]; }
  return r;
}

__device__ __forceinline__ float wave_reduce_add16(float v) {
  v += __shfl_xor(v, 1, 32);
  v += __shfl_xor(v, 2, 32);
  v += __shfl_xor(v, 4, 32);
  v += __shfl_xor(v, 8, 32);
  return v;
}
__device__ __forceinline__ float wave_reduce_add32(float v) {
  v = wave_reduce_add16(v);
  v += __shfl_xor(v, 16, 32);
  return v;
}

// ---- prep: convert W1/W2 (f32) to f16 in workspace, once per launch --------
__global__ __launch_bounds__(256) void cvt_weights_kernel(const float* __restrict__ W1,
                                                          const float* __restrict__ W2,
                                                          _Float16* __restrict__ out) {
  int i = blockIdx.x * 256 + threadIdx.x;
  if (i < WELEM) {
    out[i]         = (_Float16)W1[i];
    out[WELEM + i] = (_Float16)W2[i];
  }
}

__global__ __launch_bounds__(128) void mgan_mixer_kernel(
    const float* __restrict__ agent_qs, const float* __restrict__ states,
    const float* __restrict__ indiv_us, const float* __restrict__ adj_list,
    const _Float16* __restrict__ W1h, const _Float16* __restrict__ W2h,
    const float* __restrict__ out_w, const float* __restrict__ out_b,
    const float* __restrict__ hw_w1, const float* __restrict__ hw_b1,
    const float* __restrict__ hw_w2, const float* __restrict__ hw_b2,
    const float* __restrict__ hc_w1, const float* __restrict__ hc_b1,
    const float* __restrict__ hc_w2, const float* __restrict__ hc_b2,
    float* __restrict__ out) {
  __shared__ __attribute__((aligned(16))) _Float16 xh[NAG][LDF];            // x as f16
  __shared__ __attribute__((aligned(16))) float    adjm[NAG][NAG];          // adj + I
  __shared__ __attribute__((aligned(16))) _Float16 featsF[NHEAD][NAG][LDF]; // layer outputs
  __shared__ __attribute__((aligned(16))) _Float16 neighF[NHEAD][NAG][LDF]; // neigh
  __shared__ __attribute__((aligned(16))) _Float16 Pm[NHEAD][NAG][LDT];     // softmax probs
  __shared__ float hidW[HYPX];
  __shared__ float hidC[HYPX];
  __shared__ float headPart[NHEAD];
  __shared__ float hcS;

  const int b    = blockIdx.x;
  const int tid  = threadIdx.x;
  const int lane = tid & 31;
  const int h    = tid >> 5;   // wave == head
  const int hi   = lane >> 4;
  const int c0   = lane & 15;

  // ---- stage x (f32 -> f16) and adj(+I) into LDS -------------------------
  const float* xg = indiv_us + (size_t)b * NAG * DOBS;
#pragma unroll 4
  for (int i = tid; i < NAG * DOBS; i += 128) {
    int m = i >> 6, d = i & 63;
    xh[m][d] = (_Float16)xg[i];
  }
  const float* ag = adj_list + (size_t)b * NAG * NAG;
#pragma unroll 4
  for (int i = tid; i < NAG * NAG; i += 128) {
    int r = i >> 5, c = i & 31;
    adjm[r][c] = ag[i] + ((r == c) ? 1.0f : 0.0f);
  }
  __syncthreads();

  const _Float16* inF = &xh[0][0];
  _Float16* myF = &featsF[h][0][0];
  _Float16* myN = &neighF[h][0][0];
  _Float16* myP = &Pm[h][0][0];

  for (int layer = 0; layer < 2; ++layer) {
    const _Float16* W = (layer == 0 ? W1h : W2h) + (size_t)h * (HIDN * 128);

    // ---- attention logits: att = F F^T  (M=N=32, K=64) -------------------
    v8f att[2][2];
#pragma unroll
    for (int ti = 0; ti < 2; ++ti)
#pragma unroll
      for (int tj = 0; tj < 2; ++tj)
        att[ti][tj] = (v8f){0.f, 0.f, 0.f, 0.f, 0.f, 0.f, 0.f, 0.f};
#pragma unroll
    for (int kc = 0; kc < 2; ++kc) {
      v16h bf[2];
#pragma unroll
      for (int tj = 0; tj < 2; ++tj)
        bf[tj] = frag_B_nm(inF, LDF, tj * 16, kc * 32, lane);  // B = F^T
#pragma unroll
      for (int ti = 0; ti < 2; ++ti) {
        v16h a = frag_A(inF, LDF, ti * 16, kc * 32, lane);
#pragma unroll
        for (int tj = 0; tj < 2; ++tj) att[ti][tj] = WMMA(a, bf[tj], att[ti][tj]);
      }
    }

    // ---- masked softmax (f32), write probs as f16 ------------------------
#pragma unroll
    for (int ti = 0; ti < 2; ++ti) {
#pragma unroll
      for (int r = 0; r < 8; ++r) {
        int M = ti * 16 + r + (hi << 3);
        float a0 = att[ti][0][r];
        float a1 = att[ti][1][r];
        if (adjm[M][c0]      == 0.0f) a0 = NEGV;
        if (adjm[M][c0 + 16] == 0.0f) a1 = NEGV;
        float mx = fmaxf(a0, a1);
        mx = fmaxf(mx, __shfl_xor(mx, 1, 32));
        mx = fmaxf(mx, __shfl_xor(mx, 2, 32));
        mx = fmaxf(mx, __shfl_xor(mx, 4, 32));
        mx = fmaxf(mx, __shfl_xor(mx, 8, 32));
        float e0 = __expf(a0 - mx);
        float e1 = __expf(a1 - mx);
        float sm = wave_reduce_add16(e0 + e1);
        float inv = 1.0f / sm;
        myP[M * LDT + c0]      = (_Float16)(e0 * inv);
        myP[M * LDT + c0 + 16] = (_Float16)(e1 * inv);
      }
    }

    // ---- neigh = P @ F  (M=32, N=64, K=32) -------------------------------
    v16h bn[4];
#pragma unroll
    for (int tn = 0; tn < 4; ++tn)
      bn[tn] = frag_B_km(inF, LDF, tn * 16, 0, lane);  // B = F (k-major gather)
    v8f nb[2][4];
#pragma unroll
    for (int ti = 0; ti < 2; ++ti) {
      v16h a = frag_A(myP, LDT, ti * 16, 0, lane);
#pragma unroll
      for (int tn = 0; tn < 4; ++tn) {
        v8f z = (v8f){0.f, 0.f, 0.f, 0.f, 0.f, 0.f, 0.f, 0.f};
        nb[ti][tn] = WMMA(a, bn[tn], z);
      }
    }
#pragma unroll
    for (int ti = 0; ti < 2; ++ti)
#pragma unroll
      for (int tn = 0; tn < 4; ++tn)
#pragma unroll
        for (int r = 0; r < 8; ++r) {
          int M = ti * 16 + r + (hi << 3);
          myN[M * LDF + tn * 16 + c0] = (_Float16)nb[ti][tn][r];
        }

    // ---- linear: out = relu([F|neigh] @ W^T)  (M=32, N=64, K=128) --------
    v8f ob[2][4];
#pragma unroll
    for (int ti = 0; ti < 2; ++ti)
#pragma unroll
      for (int tn = 0; tn < 4; ++tn)
        ob[ti][tn] = (v8f){0.f, 0.f, 0.f, 0.f, 0.f, 0.f, 0.f, 0.f};
#pragma unroll
    for (int kf = 0; kf < 4; ++kf) {
      const _Float16* asrc = (kf < 2) ? inF : myN;
      int ak0 = (kf & 1) * 32;
      v16h bw[4];
#pragma unroll
      for (int tn = 0; tn < 4; ++tn) bw[tn] = frag_B_wh(W, tn * 16, kf * 32, lane);
#pragma unroll
      for (int ti = 0; ti < 2; ++ti) {
        v16h a = frag_A(asrc, LDF, ti * 16, ak0, lane);
#pragma unroll
        for (int tn = 0; tn < 4; ++tn) ob[ti][tn] = WMMA(a, bw[tn], ob[ti][tn]);
      }
    }
#pragma unroll
    for (int ti = 0; ti < 2; ++ti)
#pragma unroll
      for (int tn = 0; tn < 4; ++tn)
#pragma unroll
        for (int r = 0; r < 8; ++r) {
          int M = ti * 16 + r + (hi << 3);
          myF[M * LDF + tn * 16 + c0] = (_Float16)fmaxf(ob[ti][tn][r], 0.0f);
        }
    inF = myF;
  }

  // ---- hypernet hidden layers (cooperative, 128 threads) -----------------
  const float* sv = states + (size_t)b * SDIMX;
  {
    int j = tid & 63;
    const float* wrow = (tid < 64 ? hw_w1 : hc_w1) + (size_t)j * SDIMX;
    float acc = (tid < 64 ? hw_b1 : hc_b1)[j];
#pragma unroll 8
    for (int k = 0; k < SDIMX; ++k) acc += sv[k] * wrow[k];
    acc = fmaxf(acc, 0.0f);
    if (tid < 64) hidW[j] = acc; else hidC[j] = acc;
  }
  __syncthreads();

  // ---- ow = enc . out_w + b, softmax over agents, mix --------------------
  {
    int n = lane;
    float o = out_b[0];
    const _Float16* fr = &featsF[h][n][0];
#pragma unroll 8
    for (int e = 0; e < HIDN; ++e) o += (float)fr[e] * out_w[e];
    float mx = o;
    mx = fmaxf(mx, __shfl_xor(mx, 1, 32));
    mx = fmaxf(mx, __shfl_xor(mx, 2, 32));
    mx = fmaxf(mx, __shfl_xor(mx, 4, 32));
    mx = fmaxf(mx, __shfl_xor(mx, 8, 32));
    mx = fmaxf(mx, __shfl_xor(mx, 16, 32));
    float ex = __expf(o - mx);
    float ssum = wave_reduce_add32(ex);
    float ow = ex / ssum;
    float part = wave_reduce_add32(agent_qs[(size_t)b * NAG + n] * ow);  // qs_tot[h]
    // hw[h] = |hidden . hw_w2[h] + hw_b2[h]|
    float hp = hidW[lane] * hw_w2[h * HYPX + lane] +
               hidW[lane + 32] * hw_w2[h * HYPX + lane + 32];
    hp = wave_reduce_add32(hp);
    float hwv = fabsf(hp + hw_b2[h]);
    if (lane == 0) headPart[h] = part * hwv;
    if (h == 0) {
      float cp = hidC[lane] * hc_w2[lane] + hidC[lane + 32] * hc_w2[lane + 32];
      cp = wave_reduce_add32(cp);
      if (lane == 0) hcS = cp + hc_b2[0];
    }
  }
  __syncthreads();
  if (tid == 0) out[b] = headPart[0] + headPart[1] + headPart[2] + headPart[3] + hcS;
}

extern "C" void kernel_launch(void* const* d_in, const int* in_sizes, int n_in,
                              void* d_out, int out_size, void* d_ws, size_t ws_size,
                              hipStream_t stream) {
  const float* agent_qs = (const float*)d_in[0];
  const float* states   = (const float*)d_in[1];
  const float* indiv_us = (const float*)d_in[2];
  const float* adj_list = (const float*)d_in[3];
  const float* W1       = (const float*)d_in[4];
  const float* W2       = (const float*)d_in[5];
  const float* out_w    = (const float*)d_in[6];
  const float* out_b    = (const float*)d_in[7];
  const float* hw_w1    = (const float*)d_in[8];
  const float* hw_b1    = (const float*)d_in[9];
  const float* hw_w2    = (const float*)d_in[10];
  const float* hw_b2    = (const float*)d_in[11];
  const float* hc_w1    = (const float*)d_in[12];
  const float* hc_b1    = (const float*)d_in[13];
  const float* hc_w2    = (const float*)d_in[14];
  const float* hc_b2    = (const float*)d_in[15];
  float* out = (float*)d_out;

  _Float16* W16 = (_Float16*)d_ws;               // [2][WELEM] f16 weights
  const _Float16* W1h = W16;
  const _Float16* W2h = W16 + WELEM;

  cvt_weights_kernel<<<dim3((WELEM + 255) / 256), dim3(256), 0, stream>>>(W1, W2, W16);

  int Btot = in_sizes[0] / NAG;  // 64*128 = 8192 batch rows
  mgan_mixer_kernel<<<dim3(Btot), dim3(128), 0, stream>>>(
      agent_qs, states, indiv_us, adj_list, W1h, W2h, out_w, out_b,
      hw_w1, hw_b1, hw_w2, hw_b2, hc_w1, hc_b1, hc_w2, hc_b2, out);
}
